// ChebyKANLinear_22368189677932
// MI455X (gfx1250) — matrix operand order repaired and checked by reading, not compiled, not run on hardware
//
#include <hip/hip_runtime.h>

// ---------------------------------------------------------------------------
// ChebyKAN linear for MI455X (gfx1250, wave32, WMMA)
//   y[b,j] = sum_{i,k} T_k(tanh(x[b,i])) * C[i,j,k]
// Fused: Chebyshev basis built in-register via recurrence -> LDS (bf16),
// coeffs repacked once to bf16 N-major, GEMM via v_wmma_f32_16x16x32_bf16.
// Double-buffered LDS pipeline; B-tile staged with GLOBAL_LOAD_ASYNC_TO_LDS
// (ASYNCcnt-tracked) when the toolchain exposes the builtin.
// ---------------------------------------------------------------------------

typedef __attribute__((ext_vector_type(16))) __bf16 v16bf;
typedef __attribute__((ext_vector_type(8)))  float  v8f;

#define DIM      512
#define NK       9
#define KTOT     4608            // DIM * NK
#define BV       16384

#define BLOCK_M  128
#define BLOCK_N  128
#define ICHUNK   32
#define CHUNK_K  (ICHUNK * NK)   // 288 K-values per chunk
#define NSTEPS   (CHUNK_K / 32)  // 9 wmma k-steps per chunk (one per degree)
#define NCHUNKS  (DIM / ICHUNK)  // 16
#define RSH      (CHUNK_K + 8)   // 296 halves -> 592B row stride (16B aligned, pad kills bank conflicts)
#define BUF_HALVES ((BLOCK_M + BLOCK_N) * RSH)  // 75776 halves per buffer
#define LDS_BYTES  (2 * BUF_HALVES * 2)         // 303104 bytes (<= 320KB WGP)

#if defined(__AMDGCN__) && __has_builtin(__builtin_amdgcn_global_load_async_to_lds_b128) && __has_builtin(__builtin_amdgcn_s_wait_asynccnt)
#define USE_ASYNC_LDS 1
#else
#define USE_ASYNC_LDS 0
#endif

#if USE_ASYNC_LDS
typedef int v4i_vs __attribute__((vector_size(16)));
typedef __attribute__((address_space(1))) v4i_vs* gptr_v4i;
typedef __attribute__((address_space(3))) v4i_vs* lptr_v4i;
#endif

__device__ __forceinline__ unsigned f2bf_raw(float f) {
  unsigned u = __builtin_bit_cast(unsigned, f);
  return (u + 0x7FFFu + ((u >> 16) & 1u)) >> 16;   // round-to-nearest-even
}
__device__ __forceinline__ unsigned pack2bf(float lo, float hi) {
  return f2bf_raw(lo) | (f2bf_raw(hi) << 16);
}

// ---------------------------------------------------------------------------
// Pass 0: repack coeffs (dim, dim, 9) f32 -> Wt bf16, N-major:
//   Wt[j * 4608 + k*512 + i] = C[i][j][k]
// ---------------------------------------------------------------------------
__global__ __launch_bounds__(256) void pack_w_kernel(
    const float* __restrict__ coeffs, unsigned short* __restrict__ Wt) {
  int t = blockIdx.x * 256 + threadIdx.x;   // 0 .. 512*512-1
  int i = t & (DIM - 1);
  int j = t >> 9;
  const float* src = coeffs + ((size_t)i * DIM + j) * NK;  // 9 contiguous f32
  unsigned short* dst = Wt + (size_t)j * KTOT + i;         // stride DIM per k
#pragma unroll
  for (int k = 0; k < NK; ++k) {
    dst[(size_t)k * DIM] = (unsigned short)f2bf_raw(src[k]);
  }
}

// ---------------------------------------------------------------------------
// Fused basis + GEMM.  Grid: 512 WGs x 256 threads (8 waves, 2M x 4N).
// Each WG: 128x128 output tile. Wave tile: 64M x 32N = 4x2 wmma tiles.
// ---------------------------------------------------------------------------
__global__ __launch_bounds__(256) void cheby_fused_gemm(
    const float* __restrict__ x, const unsigned short* __restrict__ Wt,
    float* __restrict__ y) {
  extern __shared__ unsigned short smem[];

  const int tid  = threadIdx.x;
  const int lane = tid & 31;
  const int wave = tid >> 5;
  const int wm   = wave & 1;    // 2 waves in M -> offset wm*64
  const int wn   = wave >> 1;   // 4 waves in N -> offset wn*32
  const int mb   = (int)blockIdx.x >> 2;   // 128 M-blocks
  const int nb   = (int)blockIdx.x & 3;    // 4 N-blocks
  const int m0   = mb * BLOCK_M;
  const int n0   = nb * BLOCK_N;

  const int lhalf = lane >> 4;      // 0 or 1
  const int llow  = lane & 15;

  v8f acc[4][2];
#pragma unroll
  for (int mt = 0; mt < 4; ++mt)
#pragma unroll
    for (int nt = 0; nt < 2; ++nt) acc[mt][nt] = (v8f)0.0f;

  // ---- staging: fill buffer `buf` with chunk c (A via cheb VALU, B via copy)
  auto stage = [&](int c, unsigned short* As, unsigned short* Bs) {
    const int ibase = c * ICHUNK;
    // A: 128 rows x 32 i = 2048 float2-pairs, 8 per thread, recurrence in f32.
#pragma unroll
    for (int l = 0; l < 8; ++l) {
      int p   = tid + l * 256;
      int row = p >> 4;              // 0..127
      int il  = (p & 15) * 2;        // 0..30
      float2 xv = *reinterpret_cast<const float2*>(
          x + (size_t)(m0 + row) * DIM + ibase + il);
      float ta = tanhf(xv.x), tb = tanhf(xv.y);
      ta = fminf(fmaxf(ta, -1.0f + 1e-7f), 1.0f - 1e-7f);
      tb = fminf(fmaxf(tb, -1.0f + 1e-7f), 1.0f - 1e-7f);
      unsigned* dst =
          reinterpret_cast<unsigned*>(As) + row * (RSH / 2) + (il >> 1);
      float pa = 1.0f, pb = 1.0f;    // T0
      float ca = ta,   cb = tb;      // T1
      dst[0]  = 0x3F803F80u;         // bf16(1.0) x2
      dst[16] = pack2bf(ca, cb);     // k=1 -> halves offset 32 -> dword 16
#pragma unroll
      for (int k = 2; k < NK; ++k) {
        float na = __builtin_fmaf(2.0f * ta, ca, -pa);
        float nb = __builtin_fmaf(2.0f * tb, cb, -pb);
        pa = ca; ca = na; pb = cb; cb = nb;
        dst[k * 16] = pack2bf(ca, cb);
      }
    }
    // B: Wt slice -> LDS, 128 n x 36 segs of 8 halves (b128 granules).
#pragma unroll
    for (int l = 0; l < 18; ++l) {
      int q = tid + l * 256;
      int n = q / 36;
      int s = q - n * 36;            // seg: k = s>>2, il = (s&3)*8
      const unsigned short* gsrc =
          Wt + (size_t)(n0 + n) * KTOT + (s >> 2) * DIM + ibase + (s & 3) * 8;
      unsigned short* ldst = Bs + n * RSH + s * 8;
#if USE_ASYNC_LDS
      __builtin_amdgcn_global_load_async_to_lds_b128(
          (gptr_v4i)gsrc, (lptr_v4i)ldst, 0, 0);
#else
      *reinterpret_cast<uint4*>(ldst) = *reinterpret_cast<const uint4*>(gsrc);
#endif
    }
  };

  // ---- prologue: stage chunk 0 into buffer 0
  stage(0, smem, smem + BLOCK_M * RSH);
#if USE_ASYNC_LDS
  __builtin_amdgcn_s_wait_asynccnt(0);
#endif
  __syncthreads();

  for (int c = 0; c < NCHUNKS; ++c) {
    unsigned short* curBuf = smem + (c & 1) * BUF_HALVES;
    unsigned short* As = curBuf;
    unsigned short* Bs = curBuf + BLOCK_M * RSH;

    // Issue next chunk's staging into the other buffer (overlaps with WMMAs;
    // previous-iteration barrier guarantees that buffer is no longer read).
    if (c + 1 < NCHUNKS) {
      unsigned short* nxtBuf = smem + ((c + 1) & 1) * BUF_HALVES;
      stage(c + 1, nxtBuf, nxtBuf + BLOCK_M * RSH);
    }

    // ---- Compute: 9 k-steps (one per Chebyshev order), 8 WMMAs each ----
#pragma unroll
    for (int s = 0; s < NSTEPS; ++s) {
      const int kl = s * 32;

      v16bf bfrag[2];
#pragma unroll
      for (int nt = 0; nt < 2; ++nt) {
        // B 32x16: lane<16 -> col=llow, K {kl+0..7, kl+16..23}; lane>=16 -> +8
        const unsigned short* bp =
            Bs + (wn * 32 + nt * 16 + llow) * RSH + kl + lhalf * 8;
        union { v16bf v; uint4 q[2]; } u;
        u.q[0] = *reinterpret_cast<const uint4*>(bp);
        u.q[1] = *reinterpret_cast<const uint4*>(bp + 16);
        bfrag[nt] = u.v;
      }
#pragma unroll
      for (int mt = 0; mt < 4; ++mt) {
        // A 16x32: lane<16 -> row=llow, K {kl+0..7, kl+16..23}; lane>=16 -> +8
        const unsigned short* ap =
            As + (wm * 64 + mt * 16 + llow) * RSH + kl + lhalf * 8;
        union { v16bf v; uint4 q[2]; } ua;
        ua.q[0] = *reinterpret_cast<const uint4*>(ap);
        ua.q[1] = *reinterpret_cast<const uint4*>(ap + 16);
        const v16bf afrag = ua.v;
#pragma unroll
        for (int nt = 0; nt < 2; ++nt) {
          acc[mt][nt] = __builtin_amdgcn_wmma_f32_16x16x32_bf16(
              false, afrag, false, bfrag[nt], (short)0, acc[mt][nt],
              false, false);
        }
      }
    }

#if USE_ASYNC_LDS
    __builtin_amdgcn_s_wait_asynccnt(0);   // async B copies for chunk c+1 done
#endif
    __syncthreads();   // all waves done computing c / staging c+1
  }

  // ---- Epilogue: C/D layout: VGPR r -> M = r + 8*(lane>=16), N = lane&15 ----
#pragma unroll
  for (int mt = 0; mt < 4; ++mt) {
#pragma unroll
    for (int nt = 0; nt < 2; ++nt) {
      const int mbase = m0 + wm * 64 + mt * 16 + lhalf * 8;
      const int ncol  = n0 + wn * 32 + nt * 16 + llow;
#pragma unroll
      for (int r = 0; r < 8; ++r) {
        y[(size_t)(mbase + r) * DIM + ncol] = acc[mt][nt][r];
      }
    }
  }
}

// ---------------------------------------------------------------------------
extern "C" void kernel_launch(void* const* d_in, const int* in_sizes, int n_in,
                              void* d_out, int out_size, void* d_ws,
                              size_t ws_size, hipStream_t stream) {
  const float* x      = (const float*)d_in[0];  // (16384, 512) f32
  const float* coeffs = (const float*)d_in[1];  // (512, 512, 9) f32
  float* y            = (float*)d_out;          // (16384, 512) f32
  unsigned short* Wt  = (unsigned short*)d_ws;  // 4608*512 bf16 = 4.5 MB

  pack_w_kernel<<<(DIM * DIM) / 256, 256, 0, stream>>>(coeffs, Wt);

  (void)hipFuncSetAttribute((const void*)cheby_fused_gemm,
                            hipFuncAttributeMaxDynamicSharedMemorySize,
                            LDS_BYTES);
  cheby_fused_gemm<<<(BV / BLOCK_M) * (DIM / BLOCK_N), 256, LDS_BYTES,
                     stream>>>(x, Wt, y);
}
